// MultivariateCausalAttention_44856638440071
// MI455X (gfx1250) — matrix operand order repaired and checked by reading, not compiled, not run on hardware
//
#include <hip/hip_runtime.h>

// MultivariateCausalAttention for MI455X (gfx1250), wave32 + WMMA bf16 + TDM.
// ws layout: [ attn bf16 : B*H*64*64 ][ V / out_pre bf16 : B*H*N*T*D ] ~= 135 MB.
// B operands staged column-major [col][k] in LDS; fragment loads are 2x
// ds_load_b128 per lane. LDS rows padded (+8 elems) against bank conflicts.
// Pure-bf16 staging copies use tensor_load_to_lds with D# pad_enable so the
// TDM inserts the LDS row padding in hardware.

typedef __bf16 bf16;
typedef __attribute__((ext_vector_type(16))) __bf16 v16bf;
typedef __attribute__((ext_vector_type(8)))  __bf16 v8bf;
typedef __attribute__((ext_vector_type(4)))  __bf16 v4bf;
typedef __attribute__((ext_vector_type(8)))  float  v8f;
typedef __attribute__((ext_vector_type(4)))  unsigned int v4u;
typedef __attribute__((ext_vector_type(8)))  int v8i;
typedef __attribute__((ext_vector_type(4)))  int v4i;

static constexpr int Bc = 8, Hc = 8, Nc = 64, Tc = 256, Ec = 512, Dc = 64;

__device__ __forceinline__ v8f wmma_bf16(v16bf a, v16bf b, v8f c) {
  return __builtin_amdgcn_wmma_f32_16x16x32_bf16(false, a, false, b, (short)0, c,
                                                 false, false);
}

__device__ __forceinline__ unsigned lds_off(const void* p) {
  return (unsigned)(unsigned long)p;  // low 32 bits of generic addr = LDS offset
}

// TDM 2-D (or 1-D when tile_d1==0) bf16 tile load, global -> LDS, with optional
// LDS padding. stride0 in elements. pad_word = pad_en|interval|amount bits.
__device__ __forceinline__ void tdm_load_bf16(unsigned lds, const void* gptr,
                                              unsigned tile_d0, unsigned tile_d1,
                                              unsigned tensor_d0, unsigned tensor_d1,
                                              unsigned long stride0,
                                              unsigned pad_word) {
  unsigned long ga = (unsigned long)gptr;
  v4u g0 = {1u,                                    // count=1, user mode
            lds,                                   // lds_addr
            (unsigned)ga,                          // global_addr[31:0]
            (unsigned)(ga >> 32) | (2u << 30)};    // global_addr[56:32] | type=2
  v8i g1;
  g1[0] = (int)((1u << 16) | pad_word);            // data_size=2B | pad ctl
  g1[1] = (int)((tensor_d0 & 0xFFFFu) << 16);      // tensor_dim0[15:0]
  g1[2] = (int)((tensor_d0 >> 16) | ((tensor_d1 & 0xFFFFu) << 16));
  g1[3] = (int)((tensor_d1 >> 16) | (tile_d0 << 16));  // tile_dim0
  g1[4] = (int)(tile_d1 & 0xFFFFu);                // tile_dim1 (0 => 1-D)
  g1[5] = (int)(stride0 & 0xFFFFFFFFu);            // tensor_dim0_stride[31:0]
  g1[6] = (int)((stride0 >> 32) & 0xFFFFu);        // stride[47:32]
  g1[7] = 0;
  v4i z4 = {0, 0, 0, 0};
  v8i z8 = {0, 0, 0, 0, 0, 0, 0, 0};
  __builtin_amdgcn_tensor_load_to_lds(g0, g1, z4, z4, z8, 0);
}

// A: 16x32 tile at (row0,k0) from row-major [rows][ld] LDS; ld % 8 == 0.
__device__ __forceinline__ v16bf load_A(const bf16* sm, int row0, int k0, int ld) {
  int lane = threadIdx.x & 31;
  int r = lane & 15, hi = lane >> 4;
  const bf16* p = sm + (row0 + r) * ld + k0 + hi * 8;
  v8bf lo = *(const v8bf*)p;
  v8bf hh = *(const v8bf*)(p + 16);
  return __builtin_shufflevector(lo, hh, 0, 1, 2, 3, 4, 5, 6, 7,
                                 8, 9, 10, 11, 12, 13, 14, 15);
}

// B: 32x16 tile at (k0,col0) from COLUMN-major [col][ldk] LDS; ldk % 8 == 0.
__device__ __forceinline__ v16bf load_B(const bf16* smT, int k0, int col0, int ldk) {
  int lane = threadIdx.x & 31;
  int c = lane & 15, hi = lane >> 4;
  const bf16* p = smT + (col0 + c) * ldk + k0 + hi * 8;
  v8bf lo = *(const v8bf*)p;
  v8bf hh = *(const v8bf*)(p + 16);
  return __builtin_shufflevector(lo, hh, 0, 1, 2, 3, 4, 5, 6, 7,
                                 8, 9, 10, 11, 12, 13, 14, 15);
}

__device__ __forceinline__ void store_frag_bf16(bf16* dst, v8f c, int row0, int col0,
                                                long ld) {
  int lane = threadIdx.x & 31;
  int col = lane & 15, hi = lane >> 4;
#pragma unroll
  for (int r = 0; r < 8; ++r)
    dst[(long)(row0 + (hi << 3) + r) * ld + col0 + col] = (bf16)c[r];
}

__device__ __forceinline__ void store_frag_f32(float* dst, v8f c, int row0, int col0,
                                               long ld, float scale) {
  int lane = threadIdx.x & 31;
  int col = lane & 15, hi = lane >> 4;
#pragma unroll
  for (int r = 0; r < 8; ++r)
    dst[(long)(row0 + (hi << 3) + r) * ld + col0 + col] = c[r] * scale;
}

// ---- kernel 1: Q/K projection + scores + softmax -> attn (bf16) ----------
__global__ __launch_bounds__(256) void attn_kernel(const float* __restrict__ q,
                                                   const float* __restrict__ kk,
                                                   const float* __restrict__ Wq,
                                                   const float* __restrict__ Wk,
                                                   bf16* __restrict__ attn_out) {
  __shared__ alignas(16) bf16 sX[64 * 40];   // A: x chunk [row][40]
  __shared__ alignas(16) bf16 sW[64 * 40];   // B: W chunk col-major [d][40]
  __shared__ alignas(16) bf16 sQ[64 * 72];   // [i][d], A for scores
  __shared__ alignas(16) bf16 sK[64 * 72];   // [j][d], == col-major B for scores
  __shared__ float sS[64 * 64];

  const int bh = blockIdx.x;
  const int b = bh / Hc, h = bh % Hc;
  const int tid = threadIdx.x, wave = tid >> 5;
  const int mt = wave & 3;
  const int ntb = (wave >> 2) * 2;

  for (int which = 0; which < 2; ++which) {
    const float* x = which ? kk : q;
    const float* W = which ? Wk : Wq;
    bf16* dstS = which ? sK : sQ;
    v8f acc0 = {}, acc1 = {};
    for (int k0 = 0; k0 < Ec; k0 += 32) {
      for (int it = tid; it < 64 * 8; it += 256) {
        int r = it >> 3, c4 = (it & 7) * 4;
        float4 f = *(const float4*)&x[((long)b * Nc + r) * Ec + k0 + c4];
        v4bf o = {(bf16)f.x, (bf16)f.y, (bf16)f.z, (bf16)f.w};
        *(v4bf*)&sX[r * 40 + c4] = o;
      }
      for (int it = tid; it < 64 * 8; it += 256) {
        int d = it >> 3, k4 = (it & 7) * 4;
        float4 f = *(const float4*)&W[(long)(h * 64 + d) * Ec + k0 + k4];
        v4bf o = {(bf16)f.x, (bf16)f.y, (bf16)f.z, (bf16)f.w};
        *(v4bf*)&sW[d * 40 + k4] = o;
      }
      __syncthreads();
      v16bf a = load_A(sX, mt * 16, 0, 40);
      v16bf b0 = load_B(sW, 0, ntb * 16, 40);
      v16bf b1 = load_B(sW, 0, ntb * 16 + 16, 40);
      acc0 = wmma_bf16(a, b0, acc0);
      acc1 = wmma_bf16(a, b1, acc1);
      __syncthreads();
    }
    store_frag_bf16(dstS, acc0, mt * 16, ntb * 16, 72);
    store_frag_bf16(dstS, acc1, mt * 16, ntb * 16 + 16, 72);
    __syncthreads();
  }

  {  // scores = Q @ K^T / sqrt(64)
    v8f acc0 = {}, acc1 = {};
    for (int k0 = 0; k0 < 64; k0 += 32) {
      v16bf a = load_A(sQ, mt * 16, k0, 72);
      v16bf b0 = load_B(sK, k0, ntb * 16, 72);
      v16bf b1 = load_B(sK, k0, ntb * 16 + 16, 72);
      acc0 = wmma_bf16(a, b0, acc0);
      acc1 = wmma_bf16(a, b1, acc1);
    }
    store_frag_f32(sS, acc0, mt * 16, ntb * 16, 64, 0.125f);
    store_frag_f32(sS, acc1, mt * 16, ntb * 16 + 16, 64, 0.125f);
  }
  __syncthreads();

  if (tid < 64) {
    float mx = -1e30f;
    for (int j = 0; j < 64; ++j) mx = fmaxf(mx, sS[tid * 64 + j]);
    float sum = 0.f;
    for (int j = 0; j < 64; ++j) {
      float e = __expf(sS[tid * 64 + j] - mx);
      sS[tid * 64 + j] = e;
      sum += e;
    }
    float inv = 1.f / sum;
    for (int j = 0; j < 64; ++j)
      attn_out[((long)bh * 64 + tid) * 64 + j] = (bf16)(sS[tid * 64 + j] * inv);
  }
}

// ---- kernel 2: V[b,h,n] = tril(Wconv[h,n]) @ v[b,h,n]  (256x256 @ 256x64) -
__global__ __launch_bounds__(256) void conv_kernel(const float* __restrict__ Wconv,
                                                   const float* __restrict__ v,
                                                   bf16* __restrict__ Vout) {
  __shared__ alignas(16) bf16 sW[256 * 40];  // A: t x 32s chunk, masked
  __shared__ alignas(16) bf16 sB[64 * 40];   // B col-major: [e][32s]

  const int blk = blockIdx.x;
  const int n = blk % Nc, h = (blk / Nc) % Hc, b = blk / (Nc * Hc);
  const float* Wp = Wconv + ((long)(h * Nc + n) * Tc) * Tc;
  const float* vp = v + ((((long)b * Hc + h) * Nc + n) * Tc) * Dc;
  bf16* op = Vout + ((((long)b * Hc + h) * Nc + n) * Tc) * Dc;
  const int tid = threadIdx.x, wave = tid >> 5;

  v8f acc[2][4] = {};
  for (int s0 = 0; s0 < Tc; s0 += 32) {
    for (int it = tid; it < 256 * 8; it += 256) {
      int t = it >> 3, s4 = (it & 7) * 4;
      float4 f = *(const float4*)&Wp[(long)t * Tc + s0 + s4];
      int s = s0 + s4;
      v4bf o = {(bf16)(s + 0 <= t ? f.x : 0.f), (bf16)(s + 1 <= t ? f.y : 0.f),
                (bf16)(s + 2 <= t ? f.z : 0.f), (bf16)(s + 3 <= t ? f.w : 0.f)};
      *(v4bf*)&sW[t * 40 + s4] = o;
    }
    for (int it = tid; it < 8 * 64; it += 256) {
      int s4 = it >> 6, e = it & 63;  // coalesced over e
      v4bf o = {(bf16)vp[(long)(s0 + s4 * 4 + 0) * Dc + e],
                (bf16)vp[(long)(s0 + s4 * 4 + 1) * Dc + e],
                (bf16)vp[(long)(s0 + s4 * 4 + 2) * Dc + e],
                (bf16)vp[(long)(s0 + s4 * 4 + 3) * Dc + e]};
      *(v4bf*)&sB[e * 40 + s4 * 4] = o;
    }
    __syncthreads();
    if (s0 <= wave * 32 + 31) {               // causal: skip all-zero k-chunks
      v16bf bb[4];
#pragma unroll
      for (int nt = 0; nt < 4; ++nt) bb[nt] = load_B(sB, 0, nt * 16, 40);
#pragma unroll
      for (int m = 0; m < 2; ++m) {
        if (s0 <= wave * 32 + m * 16 + 15) {
          v16bf a = load_A(sW, (wave * 2 + m) * 16, 0, 40);
#pragma unroll
          for (int nt = 0; nt < 4; ++nt)
            acc[m][nt] = wmma_bf16(a, bb[nt], acc[m][nt]);
        }
      }
    }
    __syncthreads();
  }
#pragma unroll
  for (int m = 0; m < 2; ++m)
#pragma unroll
    for (int nt = 0; nt < 4; ++nt)
      store_frag_bf16(op, acc[m][nt], (wave * 2 + m) * 16, nt * 16, 64);
}

// ---- kernel 3: in-place attn apply: P[b,h,i,t,e] = sum_j attn[i,j] V[b,h,j,t,e]
__global__ __launch_bounds__(256) void apply_kernel(const bf16* __restrict__ attn,
                                                    bf16* __restrict__ Vbuf) {
  __shared__ alignas(16) bf16 sAttn[64 * 72];  // A: [i][j], padded via TDM
  __shared__ alignas(16) bf16 sVT[256 * 72];   // B col-major: [col(t,e)][j]

  const int bh = blockIdx.y;
  const int t0 = blockIdx.x * 4;
  bf16* base = Vbuf + ((long)bh * Nc) * Tc * Dc;
  const long ldRow = (long)Tc * Dc;  // 16384
  const int tid = threadIdx.x, wave = tid >> 5;

  // attn tile: dense [i][64] -> LDS [i][72] via TDM with hardware padding:
  // pad_interval = 16x8B = 128B (value 4), pad_amount = 4 dwords (value 3).
  if (wave == 0)
    tdm_load_bf16(lds_off(sAttn), attn + (long)bh * 64 * 64,
                  /*tile*/ 4096, 0, /*tensor*/ 4096, 0, /*stride*/ 4096,
                  (1u << 20) | (4u << 22) | (3u << 25));

  for (int it = tid; it < 16 * 256; it += 256) {
    int j4 = it >> 8, col = it & 255;  // coalesced over col
    const bf16* g = base + (long)t0 * Dc + col;
    v4bf o = {g[(long)(j4 * 4 + 0) * ldRow], g[(long)(j4 * 4 + 1) * ldRow],
              g[(long)(j4 * 4 + 2) * ldRow], g[(long)(j4 * 4 + 3) * ldRow]};
    *(v4bf*)&sVT[col * 72 + j4 * 4] = o;
  }
  if (wave == 0) __builtin_amdgcn_s_wait_tensorcnt(0);
  __syncthreads();

  const int m = wave & 3;            // i-tile
  const int nt0 = (wave >> 2) * 8;   // 8 column tiles
  v8f acc[8] = {};
  for (int k0 = 0; k0 < 64; k0 += 32) {
    v16bf a = load_A(sAttn, m * 16, k0, 72);
#pragma unroll
    for (int qq = 0; qq < 8; ++qq) {
      v16bf bb = load_B(sVT, k0, (nt0 + qq) * 16, 72);
      acc[qq] = wmma_bf16(a, bb, acc[qq]);
    }
  }
  // overwrite ws slab (safe: fully staged in LDS above)
#pragma unroll
  for (int qq = 0; qq < 8; ++qq)
    store_frag_bf16(base + (long)t0 * Dc, acc[qq], m * 16, (nt0 + qq) * 16, ldRow);
}

// ---- kernel 4: out[b,i,t,d] = sum_{h,e} P[b,h,i,t,e] * Wcat[d, h*64+e] ----
__global__ __launch_bounds__(256) void cat_kernel(const bf16* __restrict__ P,
                                                  const float* __restrict__ Wcat,
                                                  float* __restrict__ out) {
  __shared__ alignas(16) bf16 sA[64 * 40];  // A: [t_local][32e], padded via TDM
  __shared__ alignas(16) bf16 sB[64 * 40];  // B col-major: [d][32e]

  const int r0 = blockIdx.x * 64;     // rows share (b,i), t in [t0,t0+64)
  const int t0 = r0 % Tc;
  const int bi = r0 / Tc;
  const int i = bi % Nc, b = bi / Nc;
  const int tid = threadIdx.x, wave = tid >> 5;
  const int mt = wave & 3, ntb = (wave >> 2) * 2;

  v8f acc0 = {}, acc1 = {};
  for (int h = 0; h < Hc; ++h) {
    const bf16* Ph = P + ((((long)b * Hc + h) * Nc + i) * Tc) * Dc;
    for (int e0 = 0; e0 < 64; e0 += 32) {
      // A tile: bf16 64 rows x 32 cols, row stride 64 elems; LDS pad 32->40:
      // pad_interval = 8x8B = 64B (value 3), pad_amount = 4 dwords (value 3).
      if (wave == 0)
        tdm_load_bf16(lds_off(sA), Ph + (long)t0 * Dc + e0,
                      /*tile*/ 32, 64, /*tensor*/ 32, 64, /*stride*/ 64,
                      (1u << 20) | (3u << 22) | (3u << 25));
      for (int it = tid; it < 64 * 8; it += 256) {
        int d = it >> 3, k4 = (it & 7) * 4;
        float4 f = *(const float4*)&Wcat[(long)d * (Hc * Dc) + h * 64 + e0 + k4];
        v4bf o = {(bf16)f.x, (bf16)f.y, (bf16)f.z, (bf16)f.w};
        *(v4bf*)&sB[d * 40 + k4] = o;
      }
      if (wave == 0) __builtin_amdgcn_s_wait_tensorcnt(0);
      __syncthreads();
      v16bf a = load_A(sA, mt * 16, 0, 40);
      v16bf b0 = load_B(sB, 0, ntb * 16, 40);
      v16bf b1 = load_B(sB, 0, ntb * 16 + 16, 40);
      acc0 = wmma_bf16(a, b0, acc0);
      acc1 = wmma_bf16(a, b1, acc1);
      __syncthreads();
    }
  }
  store_frag_f32(out + (long)r0 * Dc, acc0, mt * 16, ntb * 16, Dc, 1.0f);
  store_frag_f32(out + (long)r0 * Dc, acc1, mt * 16, ntb * 16 + 16, Dc, 1.0f);
}

extern "C" void kernel_launch(void* const* d_in, const int* in_sizes, int n_in,
                              void* d_out, int out_size, void* d_ws, size_t ws_size,
                              hipStream_t stream) {
  (void)in_sizes; (void)n_in; (void)out_size; (void)ws_size;
  const float* q     = (const float*)d_in[0];
  const float* k     = (const float*)d_in[1];
  const float* v     = (const float*)d_in[2];
  const float* Wq    = (const float*)d_in[3];
  const float* Wk    = (const float*)d_in[4];
  const float* Wconv = (const float*)d_in[5];
  const float* Wcat  = (const float*)d_in[6];
  float* out = (float*)d_out;

  bf16* attn = (bf16*)d_ws;                                   // 0.5 MB
  bf16* Vbuf = attn + (long)Bc * Hc * Nc * Nc;                // 134 MB

  attn_kernel<<<Bc * Hc, 256, 0, stream>>>(q, k, Wq, Wk, attn);
  conv_kernel<<<Bc * Hc * Nc, 256, 0, stream>>>(Wconv, v, Vbuf);
  apply_kernel<<<dim3(Tc / 4, Bc * Hc), 256, 0, stream>>>(attn, Vbuf);
  cat_kernel<<<(Bc * Nc * Tc) / 64, 256, 0, stream>>>(Vbuf, Wcat, out);
}
